// MultiHeadAttention_45870250721354
// MI455X (gfx1250) — matrix operand order repaired and checked by reading, not compiled, not run on hardware
//
#include <hip/hip_runtime.h>
#include <hip/hip_bf16.h>

#define D_MODEL 768
#define NUM_HEADS 12
#define D_K 64
#define BATCH 2
#define SEQ 2048
#define M_ROWS (BATCH * SEQ)      // 4096
#define K_STEPS (D_MODEL / 32)    // 24
#define N_TILES (D_MODEL / 16)    // 48
#define M_TILES (M_ROWS / 16)     // 256
#define NG_TILES (N_TILES / 4)    // 12 groups of 4 n-tiles
#define WPB 8                     // waves per block (256 threads, wave32)

typedef __attribute__((ext_vector_type(16))) __bf16 v16bf;
typedef __attribute__((ext_vector_type(8)))  __bf16 v8bf;
typedef __attribute__((ext_vector_type(8)))  float  v8f;
typedef __attribute__((ext_vector_type(4)))  unsigned int u32x4;
typedef __attribute__((ext_vector_type(8)))  int i32x8;
typedef __attribute__((ext_vector_type(4)))  int i32x4;

#if __has_builtin(__builtin_amdgcn_tensor_load_to_lds)
#define USE_TDM 1
#else
#define USE_TDM 0
#endif

__device__ __forceinline__ v8f wmma_bf16(v16bf a, v16bf b, v8f c) {
  return __builtin_amdgcn_wmma_f32_16x16x32_bf16(false, a, false, b, (short)0, c,
                                                 false, false);
}

// A-matrix fragment (16x32 bf16). Lane m = lane&15 is row (row0+m).
// Element i -> k = k0 + i + 16*(lane>=16). Two 16B loads.
__device__ __forceinline__ v16bf load_a_frag(const __bf16* __restrict__ base,
                                             int ld, int row0, int k0) {
  int lane = threadIdx.x & 31;
  int m = lane & 15, hi = lane >> 4;
  const __bf16* p = base + (size_t)(row0 + m) * ld + k0 + hi * 16;
  v8bf c0 = *(const v8bf*)(p);
  v8bf c1 = *(const v8bf*)(p + 8);
  v16bf out;
#pragma unroll
  for (int i = 0; i < 8; ++i) { out[i] = c0[i]; out[i + 8] = c1[i]; }
  return out;
}

// B-matrix fragment (32x16 bf16). Source is row-major with B's columns as rows
// (i.e. W for X@W^T, K-tile for Q@K^T, V^T-tile for P@V).
// Lane n = lane&15 is column (col0+n); element i -> k = k0 + i + 16*(lane>=16).
__device__ __forceinline__ v16bf load_b_frag(const __bf16* __restrict__ base,
                                             int ld, int col0, int k0) {
  int lane = threadIdx.x & 31;
  int n = lane & 15, hi = lane >> 4;
  const __bf16* p = base + (size_t)(col0 + n) * ld + k0 + hi * 16;
  return *(const v16bf*)(p);
}

#if USE_TDM
// Tensor Data Mover: load a 2D tile (tile_w elems x tile_h rows, bf16) whose
// rows are stride_elems apart in global memory, packed contiguously into LDS.
// D# layout per CDNA5 ISA 8.3/8.4: group0 = {count/flags, lds_addr,
// global_addr[56:0], type=2}; group1 = {data_size=2B, tensor_dim0/1,
// tile_dim0/1, tensor_dim0_stride}. Groups 2/3 zero (2D tensor).
__device__ __forceinline__ void tdm_load_2d(unsigned lds_off,
                                            const __bf16* gptr,
                                            unsigned tile_w, unsigned tile_h,
                                            unsigned long long stride_elems) {
  unsigned long long ga = (unsigned long long)(uintptr_t)gptr;
  u32x4 g0;
  g0.x = 1u;                                           // count=1, user desc
  g0.y = lds_off;                                      // lds_addr (bytes)
  g0.z = (unsigned)ga;                                 // global_addr[31:0]
  g0.w = (unsigned)((ga >> 32) & 0x01ffffffu) | (2u << 30);  // [56:32], type=2

  unsigned long long td0 = tile_w;                     // tensor_dim0 (elems)
  unsigned long long td1 = tile_h;                     // tensor_dim1
  unsigned long long w0 = (1ull << 16)                 // data_size = 1 -> 2B
                        | ((td0 & 0xffffull) << 48);   // tensor_dim0[15:0]
  unsigned long long w1 = ((td0 >> 16) & 0xffffull)    // tensor_dim0[31:16]
                        | ((td1 & 0xffffffffull) << 16)          // tensor_dim1
                        | ((unsigned long long)tile_w << 48);    // tile_dim0
  unsigned long long w2 = (unsigned long long)tile_h             // tile_dim1
                        | ((stride_elems & 0xffffffffull) << 32);// stride[31:0]
  unsigned long long w3 = (stride_elems >> 32) & 0xffffull;      // stride[47:32]
  i32x8 g1;
  g1[0] = (int)(unsigned)w0; g1[1] = (int)(unsigned)(w0 >> 32);
  g1[2] = (int)(unsigned)w1; g1[3] = (int)(unsigned)(w1 >> 32);
  g1[4] = (int)(unsigned)w2; g1[5] = (int)(unsigned)(w2 >> 32);
  g1[6] = (int)(unsigned)w3; g1[7] = (int)(unsigned)(w3 >> 32);
  i32x4 z4 = {};
  i32x8 z8 = {};
  __builtin_amdgcn_tensor_load_to_lds(g0, g1, z4, z4, z8, 0);
}
#endif

__global__ void cvt_f32_bf16(const float* __restrict__ s, __bf16* __restrict__ d,
                             int n) {
  int i = (blockIdx.x * blockDim.x + threadIdx.x) * 4;
  if (i + 3 < n) {
    float4 f = *(const float4*)(s + i);
    d[i + 0] = (__bf16)f.x;
    d[i + 1] = (__bf16)f.y;
    d[i + 2] = (__bf16)f.z;
    d[i + 3] = (__bf16)f.w;
  } else {
    for (; i < n; ++i) d[i] = (__bf16)s[i];
  }
}

// Software-pipelined 16x64 GEMM body: prefetch next K-step's fragments into
// fresh locals while the current WMMAs issue, peel the last step.
#define GEMM_PIPELINE(X_, W_, mt_, ng_)                                       \
  v16bf a  = load_a_frag(X_, D_MODEL, (mt_) * 16, 0);                         \
  v16bf b0 = load_b_frag(W_, D_MODEL, (ng_) * 64 +  0, 0);                    \
  v16bf b1 = load_b_frag(W_, D_MODEL, (ng_) * 64 + 16, 0);                    \
  v16bf b2 = load_b_frag(W_, D_MODEL, (ng_) * 64 + 32, 0);                    \
  v16bf b3 = load_b_frag(W_, D_MODEL, (ng_) * 64 + 48, 0);                    \
  for (int ks = 0; ks < K_STEPS - 1; ++ks) {                                  \
    v16bf an = load_a_frag(X_, D_MODEL, (mt_) * 16, (ks + 1) * 32);           \
    v16bf c0 = load_b_frag(W_, D_MODEL, (ng_) * 64 +  0, (ks + 1) * 32);      \
    v16bf c1 = load_b_frag(W_, D_MODEL, (ng_) * 64 + 16, (ks + 1) * 32);      \
    v16bf c2 = load_b_frag(W_, D_MODEL, (ng_) * 64 + 32, (ks + 1) * 32);      \
    v16bf c3 = load_b_frag(W_, D_MODEL, (ng_) * 64 + 48, (ks + 1) * 32);      \
    acc0 = wmma_bf16(a, b0, acc0);                                            \
    acc1 = wmma_bf16(a, b1, acc1);                                            \
    acc2 = wmma_bf16(a, b2, acc2);                                            \
    acc3 = wmma_bf16(a, b3, acc3);                                            \
    a = an; b0 = c0; b1 = c1; b2 = c2; b3 = c3;                               \
  }                                                                           \
  acc0 = wmma_bf16(a, b0, acc0);                                              \
  acc1 = wmma_bf16(a, b1, acc1);                                              \
  acc2 = wmma_bf16(a, b2, acc2);                                              \
  acc3 = wmma_bf16(a, b3, acc3);

// C = X @ W^T (M=4096, N=768, K=768), bf16 in/out, f32 accumulate.
// mode 0: out[((b*H+h)*S+s)*64 + d] = scale*acc   (per-head Q/K layout)
// mode 1: out[((b*H+h)*64 + d)*S + s] = acc        (V transposed per head)
__global__ __launch_bounds__(256) void proj_gemm(const __bf16* __restrict__ X,
                                                 const __bf16* __restrict__ W,
                                                 __bf16* __restrict__ out,
                                                 int mode, float scale) {
  int wave = threadIdx.x >> 5;
  int tile = blockIdx.x * WPB + wave;
  if (tile >= M_TILES * NG_TILES) return;
  int mt = tile / NG_TILES, ng = tile % NG_TILES;
  int lane = threadIdx.x & 31, hi = lane >> 4;

  v8f acc0 = {}, acc1 = {}, acc2 = {}, acc3 = {};
  { GEMM_PIPELINE(X, W, mt, ng) }

  v8f accs[4] = {acc0, acc1, acc2, acc3};
#pragma unroll
  for (int j = 0; j < 4; ++j) {
    int n = ng * 64 + j * 16 + (lane & 15);
    int h = n >> 6, dd = n & 63;
#pragma unroll
    for (int r = 0; r < 8; ++r) {
      int m = mt * 16 + r + hi * 8;
      int b = m >> 11, s = m & (SEQ - 1);
      float v = accs[j][r] * scale;
      size_t idx;
      if (mode == 0)
        idx = (((size_t)b * NUM_HEADS + h) * SEQ + s) * D_K + dd;
      else
        idx = (((size_t)b * NUM_HEADS + h) * D_K + dd) * SEQ + s;
      out[idx] = (__bf16)v;
    }
  }
}

// Final projection: out_f32 = O_bf16 @ Wo^T, 16x64 per wave.
__global__ __launch_bounds__(256) void out_gemm(const __bf16* __restrict__ X,
                                                const __bf16* __restrict__ W,
                                                float* __restrict__ out) {
  int wave = threadIdx.x >> 5;
  int tile = blockIdx.x * WPB + wave;
  if (tile >= M_TILES * NG_TILES) return;
  int mt = tile / NG_TILES, ng = tile % NG_TILES;
  int lane = threadIdx.x & 31, hi = lane >> 4;

  v8f acc0 = {}, acc1 = {}, acc2 = {}, acc3 = {};
  { GEMM_PIPELINE(X, W, mt, ng) }

  v8f accs[4] = {acc0, acc1, acc2, acc3};
#pragma unroll
  for (int j = 0; j < 4; ++j) {
    int n = ng * 64 + j * 16 + (lane & 15);
#pragma unroll
    for (int r = 0; r < 8; ++r) {
      int m = mt * 16 + r + hi * 8;
      out[(size_t)m * D_MODEL + n] = accs[j][r];
    }
  }
}

// Flash attention. grid = (SEQ/128, BATCH*NUM_HEADS), block = 256.
// Wave w owns 16 query rows. K-tile / V^T-tile (64 keys) double-buffered in
// LDS; wave 0 drives the Tensor Data Mover one tile ahead (TDM ignores EXEC,
// tensor ops complete in-order per wave, so TENSORcnt<=2 means the current
// tile's two descriptors have landed). Scores pre-scaled via Q projection.
__global__ __launch_bounds__(256) void attn_kernel(const __bf16* __restrict__ Qh,
                                                   const __bf16* __restrict__ Kh,
                                                   const __bf16* __restrict__ Vt,
                                                   __bf16* __restrict__ O) {
  __shared__ __align__(64) __bf16 ktile[2][64 * 64];       // [key][d]
  __shared__ __align__(64) __bf16 vtile[2][64 * 64];       // [d][key]
  __shared__ __align__(64) __bf16 pbuf[WPB][16 * 64];      // per-wave P

  int tid = threadIdx.x, lane = tid & 31, wave = tid >> 5, hi = lane >> 4;
  int bh = blockIdx.y;
  int q0 = blockIdx.x * (16 * WPB) + wave * 16;

  const __bf16* qbase = Qh + ((size_t)bh * SEQ + q0) * D_K;
  v16bf qa0 = load_a_frag(qbase, D_K, 0, 0);
  v16bf qa1 = load_a_frag(qbase, D_K, 0, 32);

  v8f o0 = {}, o1 = {}, o2 = {}, o3 = {};
  float mrow[8], lrow[8];
#pragma unroll
  for (int r = 0; r < 8; ++r) { mrow[r] = -1e30f; lrow[r] = 0.f; }

  const __bf16* kbh = Kh + (size_t)bh * SEQ * D_K;
  const __bf16* vbh = Vt + (size_t)bh * D_K * SEQ;
  const int NT = SEQ / 64;

#if USE_TDM
  if (wave == 0) {   // pre-issue tile 0
    tdm_load_2d((unsigned)(uintptr_t)ktile[0], kbh, 64, 64, 64ull);
    tdm_load_2d((unsigned)(uintptr_t)vtile[0], vbh, 64, 64,
                (unsigned long long)SEQ);
  }
#endif

  for (int it = 0; it < NT; ++it) {
    int cur = it & 1;
#if USE_TDM
    __syncthreads();   // all waves done reading the buffer we overwrite next
    if (wave == 0) {
      if (it + 1 < NT) {
        tdm_load_2d((unsigned)(uintptr_t)ktile[cur ^ 1],
                    kbh + (size_t)(it + 1) * 64 * D_K, 64, 64, 64ull);
        tdm_load_2d((unsigned)(uintptr_t)vtile[cur ^ 1],
                    vbh + (size_t)(it + 1) * 64, 64, 64,
                    (unsigned long long)SEQ);
        __builtin_amdgcn_s_wait_tensorcnt(2);  // tile `it` complete
      } else {
        __builtin_amdgcn_s_wait_tensorcnt(0);
      }
    }
    __syncthreads();   // tile `it` visible to all waves
#else
    __syncthreads();
    {
      const uint4* src = (const uint4*)(kbh + (size_t)it * 64 * D_K);
      uint4* dst = (uint4*)ktile[cur];
      uint4* vdst = (uint4*)vtile[cur];
#pragma unroll
      for (int i = 0; i < 2; ++i) {
        int idx = tid + i * 256;
        dst[idx] = src[idx];
        int dd = idx >> 3, c = idx & 7;
        vdst[idx] = *(const uint4*)(vbh + (size_t)dd * SEQ + it * 64 + c * 8);
      }
    }
    __syncthreads();
#endif
    const __bf16* kt = ktile[cur];
    const __bf16* vt = vtile[cur];

    // scores: 16q x 64k, two K-steps over dk=64
    v8f z = {};
    v8f s0 = wmma_bf16(qa0, load_b_frag(kt, 64, 0, 0),
             wmma_bf16(qa1, load_b_frag(kt, 64, 0, 32), z));
    v8f s1 = wmma_bf16(qa0, load_b_frag(kt, 64, 16, 0),
             wmma_bf16(qa1, load_b_frag(kt, 64, 16, 32), z));
    v8f s2 = wmma_bf16(qa0, load_b_frag(kt, 64, 32, 0),
             wmma_bf16(qa1, load_b_frag(kt, 64, 32, 32), z));
    v8f s3 = wmma_bf16(qa0, load_b_frag(kt, 64, 48, 0),
             wmma_bf16(qa1, load_b_frag(kt, 64, 48, 32), z));

    // online softmax: rows live in element index r (row = r + 8*hi);
    // butterfly masks 1..8 keep the two 16-lane halves independent (wave32).
    float mnew[8];
#pragma unroll
    for (int r = 0; r < 8; ++r)
      mnew[r] = fmaxf(fmaxf(s0[r], s1[r]), fmaxf(s2[r], s3[r]));
#pragma unroll
    for (int off = 1; off < 16; off <<= 1)
#pragma unroll
      for (int r = 0; r < 8; ++r)
        mnew[r] = fmaxf(mnew[r], __shfl_xor(mnew[r], off, 32));

    float alpha[8], rs[8];
#pragma unroll
    for (int r = 0; r < 8; ++r) {
      float mn = fmaxf(mrow[r], mnew[r]);
      alpha[r] = __expf(mrow[r] - mn);
      mrow[r] = mn;
      rs[r] = 0.f;
    }

    int colb = lane & 15;
#pragma unroll
    for (int r = 0; r < 8; ++r) {
      int row = r + hi * 8;
      float p0 = __expf(s0[r] - mrow[r]); rs[r] += p0;
      float p1 = __expf(s1[r] - mrow[r]); rs[r] += p1;
      float p2 = __expf(s2[r] - mrow[r]); rs[r] += p2;
      float p3 = __expf(s3[r] - mrow[r]); rs[r] += p3;
      pbuf[wave][row * 64 +  0 + colb] = (__bf16)p0;
      pbuf[wave][row * 64 + 16 + colb] = (__bf16)p1;
      pbuf[wave][row * 64 + 32 + colb] = (__bf16)p2;
      pbuf[wave][row * 64 + 48 + colb] = (__bf16)p3;
    }
#pragma unroll
    for (int off = 1; off < 16; off <<= 1)
#pragma unroll
      for (int r = 0; r < 8; ++r)
        rs[r] += __shfl_xor(rs[r], off, 32);
#pragma unroll
    for (int r = 0; r < 8; ++r) {
      lrow[r] = lrow[r] * alpha[r] + rs[r];
      o0[r] *= alpha[r]; o1[r] *= alpha[r];
      o2[r] *= alpha[r]; o3[r] *= alpha[r];
    }

    // per-wave LDS handoff (C-layout -> A-fragment); DS ops are in-order per
    // wave, wait for the stores before the cross-lane reads.
    asm volatile("s_wait_dscnt 0" ::: "memory");
    v16bf pa0 = load_a_frag(pbuf[wave], 64, 0, 0);
    v16bf pa1 = load_a_frag(pbuf[wave], 64, 0, 32);

    o0 = wmma_bf16(pa1, load_b_frag(vt, 64, 0, 32),
         wmma_bf16(pa0, load_b_frag(vt, 64, 0, 0), o0));
    o1 = wmma_bf16(pa1, load_b_frag(vt, 64, 16, 32),
         wmma_bf16(pa0, load_b_frag(vt, 64, 16, 0), o1));
    o2 = wmma_bf16(pa1, load_b_frag(vt, 64, 32, 32),
         wmma_bf16(pa0, load_b_frag(vt, 64, 32, 0), o2));
    o3 = wmma_bf16(pa1, load_b_frag(vt, 64, 48, 32),
         wmma_bf16(pa0, load_b_frag(vt, 64, 48, 0), o3));
  }

  // epilogue: O /= l, scatter back to (B,S,D) layout for the final projection
  int b = bh / NUM_HEADS, h = bh % NUM_HEADS;
  int colb = lane & 15;
#pragma unroll
  for (int r = 0; r < 8; ++r) {
    int srow = q0 + r + hi * 8;
    float inv = 1.0f / lrow[r];
    size_t base = ((size_t)b * SEQ + srow) * D_MODEL + h * D_K;
    O[base +  0 + colb] = (__bf16)(o0[r] * inv);
    O[base + 16 + colb] = (__bf16)(o1[r] * inv);
    O[base + 32 + colb] = (__bf16)(o2[r] * inv);
    O[base + 48 + colb] = (__bf16)(o3[r] * inv);
  }
}

extern "C" void kernel_launch(void* const* d_in, const int* in_sizes, int n_in,
                              void* d_out, int out_size, void* d_ws, size_t ws_size,
                              hipStream_t stream) {
  (void)in_sizes; (void)n_in; (void)out_size; (void)ws_size;
  const float* q  = (const float*)d_in[0];
  const float* k  = (const float*)d_in[1];
  const float* v  = (const float*)d_in[2];
  const float* wq = (const float*)d_in[3];
  const float* wk = (const float*)d_in[4];
  const float* wv = (const float*)d_in[5];
  const float* wo = (const float*)d_in[6];

  size_t off = 0;
  auto bump = [&](size_t elems) {
    __bf16* p = (__bf16*)((char*)d_ws + off);
    off += (elems * sizeof(__bf16) + 255) & ~(size_t)255;
    return p;
  };
  const size_t ND = (size_t)M_ROWS * D_MODEL;   // 3,145,728 elems
  const size_t WD = (size_t)D_MODEL * D_MODEL;  //   589,824 elems
  __bf16* qb  = bump(ND);
  __bf16* kb  = bump(ND);
  __bf16* vb  = bump(ND);
  __bf16* wqb = bump(WD);
  __bf16* wkb = bump(WD);
  __bf16* wvb = bump(WD);
  __bf16* wob = bump(WD);
  __bf16* Qh  = bump(ND);  // (B,H,S,dk), pre-scaled by 1/sqrt(dk)
  __bf16* Kh  = bump(ND);  // (B,H,S,dk)
  __bf16* Vt  = bump(ND);  // (B,H,dk,S)
  __bf16* Ob  = bump(ND);  // (B,S,D)

  const int cb = 256;
  auto cgrid = [&](size_t n) { return (int)((n / 4 + cb - 1) / cb); };
  cvt_f32_bf16<<<cgrid(ND), cb, 0, stream>>>(q,  qb,  (int)ND);
  cvt_f32_bf16<<<cgrid(ND), cb, 0, stream>>>(k,  kb,  (int)ND);
  cvt_f32_bf16<<<cgrid(ND), cb, 0, stream>>>(v,  vb,  (int)ND);
  cvt_f32_bf16<<<cgrid(WD), cb, 0, stream>>>(wq, wqb, (int)WD);
  cvt_f32_bf16<<<cgrid(WD), cb, 0, stream>>>(wk, wkb, (int)WD);
  cvt_f32_bf16<<<cgrid(WD), cb, 0, stream>>>(wv, wvb, (int)WD);
  cvt_f32_bf16<<<cgrid(WD), cb, 0, stream>>>(wo, wob, (int)WD);

  const int gemmBlocks = (M_TILES * NG_TILES) / WPB;  // 384
  proj_gemm<<<gemmBlocks, 256, 0, stream>>>(qb, wqb, Qh, 0, 0.125f);
  proj_gemm<<<gemmBlocks, 256, 0, stream>>>(kb, wkb, Kh, 0, 1.0f);
  proj_gemm<<<gemmBlocks, 256, 0, stream>>>(vb, wvb, Vt, 1, 1.0f);

  dim3 agrid(SEQ / (16 * WPB), BATCH * NUM_HEADS);
  attn_kernel<<<agrid, 256, 0, stream>>>(Qh, Kh, Vt, Ob);

  out_gemm<<<gemmBlocks, 256, 0, stream>>>(Ob, wob, (float*)d_out);
}
